// SimpleGenerativeLALRNN_36498632081940
// MI455X (gfx1250) — matrix operand order repaired and checked by reading, not compiled, not run on hardware
//
#include <hip/hip_runtime.h>
#include <hip/hip_bf16.h>
#include <math.h>

typedef __attribute__((ext_vector_type(16))) _Float16 v16h;
typedef __attribute__((ext_vector_type(8)))  _Float16 v8h;
typedef __attribute__((ext_vector_type(8)))  float    v8f;
typedef __attribute__((ext_vector_type(4)))  float    v4f;

#define N_LATENT 400
#define KPAD     416      // 13 * 32 (K of GEMM1)
#define KSTEPS   13
#define HPAD     416      // padded hidden width: 26 column tiles
#define CTILES   13       // column tiles per wave (half of 26)
#define N_OUT    32
#define VOCAB    64

#define SL1 40            // LDS row stride (halfs) for W1 k-slice (bank-conflict pad, 16B aligned)
#define SL2 424           // LDS row stride (halfs) for W2 / h scratch

#if defined(__has_builtin)
#if __has_builtin(__builtin_amdgcn_tanhf)
#define TANHF(v) __builtin_amdgcn_tanhf(v)
#endif
#if __has_builtin(__builtin_amdgcn_load_to_lds)
#define HAVE_LOAD_TO_LDS 1
#endif
#endif
#ifndef TANHF
#define TANHF(v) tanhf(v)
#endif

// 16B global -> LDS copy; async DMA (ASYNCcnt) when available.
__device__ __forceinline__ void lds_dma16(const _Float16* src, _Float16* dst) {
#ifdef HAVE_LOAD_TO_LDS
  __builtin_amdgcn_load_to_lds(
      (__attribute__((address_space(1))) const void*)(const void*)src,
      (__attribute__((address_space(3)))       void*)(void*)dst,
      16, 0, 0);
#else
  *(v8h*)dst = *(const v8h*)src;
#endif
}

__device__ __forceinline__ void lds_dma_fence() {
#ifdef HAVE_LOAD_TO_LDS
  asm volatile("s_wait_asynccnt 0" ::: "memory");
#endif
}

// ---------------------------------------------------------------------------
// Prep: W1 [400x400] f32 -> W1h [416x416] f16 (zero pad rows+cols),
//       W2 [32x400]  f32 -> W2h [32x416]  f16 (zero pad cols).
// ---------------------------------------------------------------------------
__global__ __launch_bounds__(256) void prep_weights(const float* __restrict__ W1,
                                                    const float* __restrict__ W2,
                                                    _Float16* __restrict__ W1h,
                                                    _Float16* __restrict__ W2h) {
  int i = blockIdx.x * 256 + threadIdx.x;
  const int t1 = HPAD * KPAD;
  const int t2 = N_OUT * KPAD;
  if (i < t1) {
    int r = i / KPAD, c = i - r * KPAD;
    W1h[i] = (r < N_LATENT && c < N_LATENT) ? (_Float16)W1[r * N_LATENT + c] : (_Float16)0.0f;
  } else if (i < t1 + t2) {
    int j = i - t1;
    int r = j / KPAD, c = j - r * KPAD;
    W2h[j] = (c < N_LATENT) ? (_Float16)W2[r * N_LATENT + c] : (_Float16)0.0f;
  }
}

// ---------------------------------------------------------------------------
// Fused MLP. 256 threads = 8 wave32 = 4 row-pairs x 2 col-halves; 128 rows/WG.
// GEMM1: wave (rp,ch) -> rows rp*32..+31, hidden cols ch*208..+207
//        (2 row-tiles x 13 col-tiles = 208 accum VGPRs, B reused 2x).
// GEMM2: wave (rp,s)  -> 16-row slice of shared h, full K=416.
// ---------------------------------------------------------------------------
__global__ __launch_bounds__(256) void mlp_kernel(
    const float*    __restrict__ x,
    const _Float16* __restrict__ W1h,
    const _Float16* __restrict__ W2h,
    const float*    __restrict__ b1,
    const float*    __restrict__ b2,
    const int*      __restrict__ idx,
    float*          __restrict__ out) {

  __shared__ __align__(16) _Float16 sW1s[2 * HPAD * SL1];  //  66,560 B: double-buffered k-slice
  __shared__ __align__(16) _Float16 sW2 [N_OUT * SL2];     //  27,136 B: full W2 (f16)
  __shared__ __align__(16) _Float16 sH  [4 * 32 * SL2];    // 108,544 B: h scratch (per row-pair)
  __shared__ __align__(16) float    sOut[8 * 16 * VOCAB];  //  32,768 B: per-wave out stage

  const int tid  = threadIdx.x;
  const int lane = tid & 31;
  const int wave = tid >> 5;
  const int rp   = wave >> 1;     // row-pair 0..3
  const int ch   = wave & 1;      // col-half (GEMM1) / row-subtile (GEMM2)
  const int m    = lane & 15;
  const int hi   = lane >> 4;     // half-wave K-interleave select
  const long RB  = (long)blockIdx.x * 128;
  const long Rw  = RB + rp * 32;  // GEMM1 row base of this wave (2 tiles)

  // Stage W2 (f16) into LDS once.
  for (int c = tid; c < N_OUT * (KPAD / 8); c += 256) {
    int r = c / (KPAD / 8);
    int g = c - r * (KPAD / 8);
    *(v8h*)&sW2[r * SL2 + g * 8] = *(const v8h*)(W2h + r * KPAD + g * 8);
  }

  // Cooperative stage of W1h[:, k0:k0+32] into slice buffer `buf`.
  auto stage = [&](int ks, int buf) {
    _Float16* dstb = sW1s + buf * (HPAD * SL1);
    const _Float16* srcb = W1h + ks * 32;
    for (int c = tid; c < HPAD * 4; c += 256) {
      int r = c >> 2, g = c & 3;
      lds_dma16(srcb + r * KPAD + g * 8, dstb + r * SL1 + g * 8);
    }
  };

  // ---- GEMM1 ---------------------------------------------------------------
  v8f acc0[CTILES], acc1[CTILES];
  #pragma unroll
  for (int t = 0; t < CTILES; ++t) { acc0[t] = {}; acc1[t] = {}; }

  const float* xrow0 = x + (Rw + m) * (long)N_LATENT;
  const float* xrow1 = xrow0 + 16 * (long)N_LATENT;

  stage(0, 0);
  lds_dma_fence();
  __syncthreads();

  #pragma unroll
  for (int ks = 0; ks < KSTEPS; ++ks) {
    const int k0 = ks * 32;

    if (ks + 1 < KSTEPS) stage(ks + 1, (ks + 1) & 1);  // async DMA overlaps compute below

    // A fragments (16x32 f16): lane<16 K {0..7,16..23}, lane>=16 K {8..15,24..31}.
    v16h a0, a1;
    {
      const int kbase = k0 + hi * 8;
      v4f f0 = *(const v4f*)(xrow0 + kbase);
      v4f f1 = *(const v4f*)(xrow0 + kbase + 4);
      v4f g0 = *(const v4f*)(xrow1 + kbase);
      v4f g1 = *(const v4f*)(xrow1 + kbase + 4);
      #pragma unroll
      for (int i2 = 0; i2 < 4; ++i2) {
        a0[i2] = (_Float16)f0[i2]; a0[4 + i2] = (_Float16)f1[i2];
        a1[i2] = (_Float16)g0[i2]; a1[4 + i2] = (_Float16)g1[i2];
      }
      if (ks < KSTEPS - 1) {
        v4f f2 = *(const v4f*)(xrow0 + kbase + 16);
        v4f f3 = *(const v4f*)(xrow0 + kbase + 20);
        v4f g2 = *(const v4f*)(xrow1 + kbase + 16);
        v4f g3 = *(const v4f*)(xrow1 + kbase + 20);
        #pragma unroll
        for (int i2 = 0; i2 < 4; ++i2) {
          a0[8 + i2] = (_Float16)f2[i2]; a0[12 + i2] = (_Float16)f3[i2];
          a1[8 + i2] = (_Float16)g2[i2]; a1[12 + i2] = (_Float16)g3[i2];
        }
        __builtin_prefetch(xrow0 + k0 + 32, 0, 0);
        __builtin_prefetch(xrow1 + k0 + 32, 0, 0);
      } else {  // K tail 400..415 is zero pad
        #pragma unroll
        for (int i2 = 0; i2 < 8; ++i2) { a0[8 + i2] = (_Float16)0.0f; a1[8 + i2] = (_Float16)0.0f; }
      }
    }

    // 13 column tiles; each B fragment feeds two WMMAs (rows tile 0 and 1).
    const _Float16* Bb = sW1s + (ks & 1) * (HPAD * SL1);
    #pragma unroll
    for (int t = 0; t < CTILES; ++t) {
      const int row = ch * 208 + t * 16 + m;   // W1h row == hidden column
      v8h b0 = *(const v8h*)&Bb[row * SL1 + hi * 8];
      v8h b1v = *(const v8h*)&Bb[row * SL1 + 16 + hi * 8];
      v16h b;
      #pragma unroll
      for (int i2 = 0; i2 < 8; ++i2) { b[i2] = b0[i2]; b[8 + i2] = b1v[i2]; }
      acc0[t] = __builtin_amdgcn_wmma_f32_16x16x32_f16(false, a0, false, b, (short)0, acc0[t], false, false);
      acc1[t] = __builtin_amdgcn_wmma_f32_16x16x32_f16(false, a1, false, b, (short)0, acc1[t], false, false);
    }

    lds_dma_fence();   // waits this wave's async stage of the next slice
    __syncthreads();
  }

  // ---- bias + tanh -> shared h scratch (pad cols come out as tanh(0)=0) ----
  _Float16* myHw = sH + rp * 32 * SL2;
  #pragma unroll
  for (int t = 0; t < CTILES; ++t) {
    const int col = ch * 208 + t * 16 + m;
    const float bb = (col < N_LATENT) ? b1[col] : 0.0f;
    #pragma unroll
    for (int j = 0; j < 8; ++j) {
      myHw[(hi * 8 + j) * SL2 + col]      = (_Float16)TANHF(acc0[t][j] + bb);
      myHw[(16 + hi * 8 + j) * SL2 + col] = (_Float16)TANHF(acc1[t][j] + bb);
    }
  }
  __syncthreads();   // h fully visible to the wave pair

  // ---- GEMM2: wave (rp, s=ch) takes 16-row slice, full K=416 ---------------
  const _Float16* myH = sH + (rp * 32 + ch * 16) * SL2;
  v8f o0 = {}, o1 = {};
  #pragma unroll
  for (int ks = 0; ks < KSTEPS; ++ks) {
    const int k0 = ks * 32;
    v16h a;
    {
      v8h a0v = *(const v8h*)&myH[m * SL2 + k0 + hi * 8];
      v8h a1v = *(const v8h*)&myH[m * SL2 + k0 + 16 + hi * 8];
      #pragma unroll
      for (int i2 = 0; i2 < 8; ++i2) { a[i2] = a0v[i2]; a[8 + i2] = a1v[i2]; }
    }
    v16h bA, bB;
    {
      v8h t0 = *(const v8h*)&sW2[m * SL2 + k0 + hi * 8];
      v8h t1 = *(const v8h*)&sW2[m * SL2 + k0 + 16 + hi * 8];
      v8h t2 = *(const v8h*)&sW2[(16 + m) * SL2 + k0 + hi * 8];
      v8h t3 = *(const v8h*)&sW2[(16 + m) * SL2 + k0 + 16 + hi * 8];
      #pragma unroll
      for (int i2 = 0; i2 < 8; ++i2) {
        bA[i2] = t0[i2]; bA[8 + i2] = t1[i2];
        bB[i2] = t2[i2]; bB[8 + i2] = t3[i2];
      }
    }
    o0 = __builtin_amdgcn_wmma_f32_16x16x32_f16(false, a, false, bA, (short)0, o0, false, false);
    o1 = __builtin_amdgcn_wmma_f32_16x16x32_f16(false, a, false, bB, (short)0, o1, false, false);
  }

  // ---- bias + softmax (half-wave shuffles) + vocab scatter -----------------
  const float bb0 = b2[m];
  const float bb1 = b2[16 + m];
  const int ic0 = idx[m];
  const int ic1 = idx[16 + m];
  float* myO = sOut + wave * 16 * VOCAB;

  {  // zero the 16x64 stage
    v4f z = {};
    #pragma unroll
    for (int i2 = 0; i2 < 8; ++i2)
      *(v4f*)&myO[(lane >> 1) * VOCAB + (lane & 1) * 32 + i2 * 4] = z;
  }

  #pragma unroll
  for (int j = 0; j < 8; ++j) {   // C layout: VGPR j -> row (hi*8+j), col = m (+16 for o1)
    float l0 = o0[j] + bb0;
    float l1 = o1[j] + bb1;
    float mx = fmaxf(l0, l1);
    #pragma unroll
    for (int off2 = 1; off2 < 16; off2 <<= 1)
      mx = fmaxf(mx, __shfl_xor(mx, off2, 32));
    float e0 = __expf(l0 - mx);
    float e1 = __expf(l1 - mx);
    float s = e0 + e1;
    #pragma unroll
    for (int off2 = 1; off2 < 16; off2 <<= 1)
      s += __shfl_xor(s, off2, 32);
    float inv = __builtin_amdgcn_rcpf(s);
    const int row = hi * 8 + j;
    myO[row * VOCAB + ic0] = e0 * inv;
    myO[row * VOCAB + ic1] = e1 * inv;
  }

  // Contiguous b128 stores of vocab cols 0..15 and 32..47 of each row.
  {
    const int r = lane >> 1, half = lane & 1;
    const long orow = RB + rp * 32 + ch * 16 + r;
    float* op = out + orow * VOCAB + half * 32;
    #pragma unroll
    for (int i2 = 0; i2 < 4; ++i2)
      *(v4f*)(op + i2 * 4) = *(const v4f*)&myO[r * VOCAB + half * 32 + i2 * 4];
  }
}

// ---------------------------------------------------------------------------
extern "C" void kernel_launch(void* const* d_in, const int* in_sizes, int n_in,
                              void* d_out, int out_size, void* d_ws, size_t ws_size,
                              hipStream_t stream) {
  const float* x  = (const float*)d_in[0];
  const float* W1 = (const float*)d_in[1];
  const float* b1 = (const float*)d_in[2];
  const float* W2 = (const float*)d_in[3];
  const float* b2 = (const float*)d_in[4];
  const int*  idx = (const int*)d_in[5];
  float* out = (float*)d_out;

  _Float16* W1h = (_Float16*)d_ws;                                    // 416*416*2 = 346,112 B
  _Float16* W2h = (_Float16*)((char*)d_ws + (size_t)HPAD * KPAD * sizeof(_Float16));

  const int nrows = in_sizes[0] / N_LATENT;   // 262144

  const int prepTot = (HPAD + N_OUT) * KPAD;
  prep_weights<<<(prepTot + 255) / 256, 256, 0, stream>>>(W1, W2, W1h, W2h);
  mlp_kernel<<<nrows / 128, 256, 0, stream>>>(x, W1h, W2h, b1, b2, idx, out);
}
